// MFVI_15951508537737
// MI455X (gfx1250) — matrix operand order repaired and checked by reading, not compiled, not run on hardware
//
#include <hip/hip_runtime.h>
#include <stdint.h>

namespace {
constexpr int BS = 8;
constexpr int NE = 80;
constexpr int D  = 768;   // repr dim (K of first GEMMs)
constexpr int R  = 768;   // CPD rank (N of first GEMMs, K of second)
constexpr int RE = 8;
constexpr int RR = 12;
constexpr int ITERS = 3;
constexpr float NEGV = -10000.0f;
constexpr int M1 = BS * NE;        // 640
constexpr int M2 = BS * NE * NE;   // 51200
}

typedef __attribute__((ext_vector_type(16))) __bf16 v16bf;
typedef __attribute__((ext_vector_type(8)))  float  v8f;
typedef __attribute__((ext_vector_type(4)))  uint32_t v4u;
typedef __attribute__((ext_vector_type(8)))  int      v8i;
typedef __attribute__((ext_vector_type(4)))  int      v4i;

union BFrag { v16bf v; uint32_t u[8]; };

#if defined(__has_builtin)
#  if __has_builtin(__builtin_amdgcn_tensor_load_to_lds)
#    define HAVE_TDM 1
#  endif
#endif
#ifndef HAVE_TDM
#  define HAVE_TDM 0
#endif

__device__ __forceinline__ uint32_t f32_to_bf16u(float f) {
  union { float f; uint32_t u; } x; x.f = f;
  uint32_t r = x.u + 0x7FFFu + ((x.u >> 16) & 1u);   // round-to-nearest-even
  return r >> 16;
}
__device__ __forceinline__ uint32_t pack2bf(float lo, float hi) {
  return f32_to_bf16u(lo) | (f32_to_bf16u(hi) << 16);
}

#if HAVE_TDM
// 1-D TDM copy of one contiguous 4 KB packed-B tile (512 x 8-byte units) into
// LDS.  D# group0: count=1, lds_addr, global_addr, type=2.  Group1: mask=0,
// data_size=3 (8B), tensor_dim0=tile_dim0=512, tensor_dim1=tile_dim1=1.
// Groups 2/3 zero (<=2D tensor).  Tracked by TENSORcnt.
__device__ __forceinline__ void tdm_load_b(const uint32_t* __restrict__ g,
                                           uint32_t* l) {
  uint64_t ga = (uint64_t)(uintptr_t)g;
  uint32_t la = (uint32_t)(uintptr_t)l;     // low 32 bits = LDS byte offset
  v4u g0;
  g0.x = 1u;                                 // count = 1 valid descriptor
  g0.y = la;                                 // lds_addr
  g0.z = (uint32_t)ga;                       // global_addr[31:0]
  g0.w = (uint32_t)(ga >> 32) | 0x80000000u; // global_addr[56:32] | type=2
  v8i g1 = { (int)0x00030000u,      // workgroup_mask=0, data_size=3 (8B)
             (int)(512u << 16),     // tensor_dim0 = 512 (low 16 in [31:16])
             (int)0x00010000u,      // tensor_dim0 hi=0 | tensor_dim1 = 1
             (int)(512u << 16),     // tensor_dim1 hi=0 | tile_dim0 = 512
             1,                     // tile_dim1 = 1 | tile_dim2 = 0
             512, 0, 0 };           // tensor_dim0_stride = 512
  v4i z4 = {0, 0, 0, 0};
#if __clang_major__ >= 23
  v8i z8 = {0, 0, 0, 0, 0, 0, 0, 0};
  __builtin_amdgcn_tensor_load_to_lds(g0, g1, z4, z4, z8, 0);
#else
  __builtin_amdgcn_tensor_load_to_lds(g0, g1, z4, z4, 0);
#endif
}
#endif

// ---------------------------------------------------------------------------
// Pack a row-major f32 [768,768] weight matrix into the WMMA B-fragment
// layout for v_wmma_f32_16x16x32_bf16:
//   Bp[((kk*N + n)<<4) + (h<<3) + p] packs K = kk*32 + h*16 + 2p (lo) / +1 (hi)
//   for column n.  Lane L = (n%16) + 16*h reads 8 contiguous dwords, and the
//   per-(block,K-step) tile of 64 columns is one contiguous 4 KB chunk (TDM).
// ---------------------------------------------------------------------------
__global__ void pack_w(const float* __restrict__ B, uint32_t* __restrict__ Bp) {
  int o = blockIdx.x * blockDim.x + threadIdx.x;
  constexpr int TOT = (768 / 32) * 768 * 16;   // 294912
  if (o >= TOT) return;
  int p  = o & 7;
  int h  = (o >> 3) & 1;
  int n  = (o >> 4) % 768;
  int kk = (o >> 4) / 768;
  int k0 = kk * 32 + h * 16 + 2 * p;
  Bp[o] = pack2bf(B[(size_t)k0 * 768 + n], B[(size_t)(k0 + 1) * 768 + n]);
}

// ---------------------------------------------------------------------------
// WMMA GEMM: C[M,N] = A[M,K] @ Wpacked + bias.  64x64 tile per 128-thread
// block (4 waves x 16 rows x 64 cols), K stepped by 32 with
// v_wmma_f32_16x16x32_bf16.  B tiles are staged into LDS with a TDM double
// buffer: wave 0 DMAs K-step kk+1 while all waves compute kk from LDS.  All
// four B fragments are pulled into registers first so the four WMMAs of a
// K-step issue back-to-back behind a single s_wait_dscnt.
// EPI 0: f32 out (L1/L2).  EPI 1: T = (..)*l1*l2 -> bf16.  EPI 2: J masked f32.
// AF32: A is f32 (converted to bf16 in-register) vs bf16.
// ---------------------------------------------------------------------------
template<int EPI, bool AF32>
__global__ void __launch_bounds__(128)
gemm_wmma(const void* __restrict__ Aptr, const uint32_t* __restrict__ Bp,
          const float* __restrict__ bias, float* __restrict__ Cf,
          uint16_t* __restrict__ Cb, const float* __restrict__ L1,
          const float* __restrict__ L2, const int* __restrict__ ent,
          int K, int N)
{
  __shared__ uint32_t sB[2][1024];   // 2 x 4 KB double buffer

  const int lane = threadIdx.x & 31;
  const int wave = threadIdx.x >> 5;
  const int m16  = lane & 15;
  const int h    = lane >> 4;
  const int rowA    = blockIdx.x * 64 + wave * 16 + m16;
  const int colBase = blockIdx.y * 64;

  const float*    Af = (const float*)Aptr;
  const uint16_t* Ah = (const uint16_t*)Aptr;

  v8f acc[4] = {};

  const int nk = K >> 5;
  // prologue: stage B tile for kk = 0
#if HAVE_TDM
  if (threadIdx.x < 32)
    tdm_load_b(Bp + (((size_t)0 * N + colBase) << 4), &sB[0][0]);
#else
  {
    const uint32_t* src = Bp + (((size_t)0 * N + colBase) << 4);
    for (int x = threadIdx.x; x < 1024; x += 128) sB[0][x] = src[x];
  }
#endif

  for (int kk = 0; kk < nk; ++kk) {
    __syncthreads();   // everyone done reading the buffer we are about to fill
#if HAVE_TDM
    if (threadIdx.x < 32) {
      if (kk + 1 < nk) {
        tdm_load_b(Bp + (((size_t)(kk + 1) * N + colBase) << 4),
                   &sB[(kk + 1) & 1][0]);
        __builtin_amdgcn_s_wait_tensorcnt((short)1);   // tile kk landed
      } else {
        __builtin_amdgcn_s_wait_tensorcnt((short)0);   // last tile landed
      }
    }
#else
    if (kk + 1 < nk) {
      const uint32_t* src = Bp + (((size_t)(kk + 1) * N + colBase) << 4);
      for (int x = threadIdx.x; x < 1024; x += 128) sB[(kk + 1) & 1][x] = src[x];
    }
#endif
    __syncthreads();   // publish tile kk to all waves

    // A fragment (16x32 bf16): lanes 0-15 rows, VGPR p -> K pairs
    //   K = kb+2p, kb = kk*32 + h*16
    BFrag a;
    const int kb = kk * 32 + h * 16;
    if (AF32) {
      const float* Ar = Af + (size_t)rowA * K;
      #pragma unroll
      for (int p = 0; p < 4; ++p) {
        float2 x = *(const float2*)(Ar + kb + 2 * p);
        a.u[p] = pack2bf(x.x, x.y);
      }
      #pragma unroll
      for (int p = 0; p < 4; ++p) {
        float2 x = *(const float2*)(Ar + kb + 8 + 2 * p);
        a.u[4 + p] = pack2bf(x.x, x.y);
      }
    } else {
      const uint16_t* Ar = Ah + (size_t)rowA * K;
      #pragma unroll
      for (int p = 0; p < 4; ++p)
        a.u[p] = *(const uint32_t*)(Ar + kb + 2 * p);
      #pragma unroll
      for (int p = 0; p < 4; ++p)
        a.u[4 + p] = *(const uint32_t*)(Ar + kb + 8 + 2 * p);
    }

    // Pull all four B fragments first (one ds_load clause, one wait), then
    // issue the four WMMAs back-to-back.
    const uint32_t* bufB = sB[kk & 1];
    BFrag b[4];
    #pragma unroll
    for (int t = 0; t < 4; ++t) {
      const uint32_t* bp = bufB + (((t * 16 + m16) << 4) + (h << 3));
      #pragma unroll
      for (int p = 0; p < 8; ++p) b[t].u[p] = bp[p];
    }
    #pragma unroll
    for (int t = 0; t < 4; ++t) {
      acc[t] = __builtin_amdgcn_wmma_f32_16x16x32_bf16(
          false, a.v, false, b[t].v, (short)0, acc[t], false, false);
    }
  }

  // D layout: VGPR p holds row p (lanes 0-15) / p+8 (lanes 16-31); col = lane%16
  #pragma unroll
  for (int p = 0; p < 8; ++p) {
    const int row = blockIdx.x * 64 + wave * 16 + h * 8 + p;
    #pragma unroll
    for (int t = 0; t < 4; ++t) {
      const int col = colBase + t * 16 + m16;
      float v = acc[t][p] + bias[col];
      if (EPI == 0) {
        Cf[(size_t)row * N + col] = v;
      } else if (EPI == 1) {
        const int b_  = row / (NE * NE);
        const int r2 = row - b_ * NE * NE;
        const int i  = r2 / NE;
        const int j  = r2 - i * NE;
        float tv = v * L1[(size_t)(b_ * NE + i) * R + col]
                     * L2[(size_t)(b_ * NE + j) * R + col];
        Cb[(size_t)row * N + col] = (uint16_t)f32_to_bf16u(tv);
      } else {
        const int b_  = row / (NE * NE);
        const int r2 = row - b_ * NE * NE;
        const int i  = r2 / NE;
        const int j  = r2 - i * NE;
        const int en = ent[b_];
        Cf[(size_t)row * N + col] = (i < en && j < en) ? v : 0.0f;
      }
    }
  }
}

// ---------------------------------------------------------------------------
// Iteration kernels
// ---------------------------------------------------------------------------
__global__ void mask1d_k(float* __restrict__ q, const int* __restrict__ ent) {
  int idx = blockIdx.x * blockDim.x + threadIdx.x;
  if (idx >= M1 * RE) return;
  int b = idx / (NE * RE);
  int n = (idx / RE) % NE;
  if (n >= ent[b]) q[idx] = NEGV;
}

__global__ void mask2d_k(float* __restrict__ q, const int* __restrict__ ent) {
  int idx = blockIdx.x * blockDim.x + threadIdx.x;
  if (idx >= M2 * RR) return;
  int pr = idx / RR;
  int b  = pr / (NE * NE);
  int r2 = pr - b * NE * NE;
  int i  = r2 / NE;
  int j  = r2 - i * NE;
  int en = ent[b];
  if (i >= en || j >= en) q[idx] = NEGV;
}

template<int C>
__global__ void softmax_rows(const float* __restrict__ qv,
                             float* __restrict__ q, int rows) {
  int r = blockIdx.x * blockDim.x + threadIdx.x;
  if (r >= rows) return;
  float v[C];
  float m = -3.4e38f;
  #pragma unroll
  for (int c = 0; c < C; ++c) { v[c] = qv[(size_t)r * C + c]; m = fmaxf(m, v[c]); }
  float s = 0.f;
  #pragma unroll
  for (int c = 0; c < C; ++c) { v[c] = __expf(v[c] - m); s += v[c]; }
  float inv = 1.f / s;
  #pragma unroll
  for (int c = 0; c < C; ++c) q[(size_t)r * C + c] = v[c] * inv;
}

// One block per (b,i,j) pair.  Stages the 768-float J row in LDS once, emits
// Fso[pair][s*8+e] and adds Fr directly into qrv.  J is zero for masked pairs,
// so masked contributions vanish naturally.
__global__ void __launch_bounds__(64)
pair_kernel(const float* __restrict__ J, const float* __restrict__ qs,
            const float* __restrict__ qo, const float* __restrict__ qr,
            float* __restrict__ Fso, float* __restrict__ qrv)
{
  __shared__ float sJ[RE * RE * RR];   // 768 floats
  __shared__ float sQs[RE], sQo[RE], sQr[RR];
  const int pair = blockIdx.x;
  const int b  = pair / (NE * NE);
  const int r2 = pair - b * NE * NE;
  const int i  = r2 / NE;
  const int j  = r2 - i * NE;
  const int t  = threadIdx.x;
  const float* Jrow = J + (size_t)pair * (RE * RE * RR);
  for (int x = t; x < RE * RE * RR; x += 64) sJ[x] = Jrow[x];
  if (t < RE) { sQs[t] = qs[(b * NE + i) * RE + t];
                sQo[t] = qo[(b * NE + j) * RE + t]; }
  if (t < RR) sQr[t] = qr[(size_t)pair * RR + t];
  __syncthreads();
  const int s = t >> 3, e = t & 7;
  const float* Jse = sJ + (s * RE + e) * RR;
  float fso = 0.f;
  #pragma unroll
  for (int r = 0; r < RR; ++r) fso += Jse[r] * sQr[r];
  Fso[(size_t)pair * 64 + t] = fso;
  if (t < RR) {
    float fr = 0.f;
    #pragma unroll 4
    for (int se = 0; se < 64; ++se)
      fr += sJ[se * RR + t] * sQs[se >> 3] * sQo[se & 7];
    qrv[(size_t)pair * RR + t] += fr;
  }
}

// Fs[b,i,s] = sum_{j,e} Fso[b,i,j,s,e] * qo[b,j,e]; accumulate into qsv.
__global__ void __launch_bounds__(64)
reduce_Fs(const float* __restrict__ Fso, const float* __restrict__ qo,
          float* __restrict__ qsv)
{
  const int bi = blockIdx.x;          // b*NE + i
  const int b  = bi / NE;
  const int t  = threadIdx.x;
  const int s  = t & 7;
  const int jg = t >> 3;
  __shared__ float part[64];
  float acc = 0.f;
  for (int j = jg; j < NE; j += 8) {
    const float* f  = Fso + ((size_t)bi * NE + j) * 64 + s * 8;
    const float* qj = qo + (b * NE + j) * 8;
    #pragma unroll
    for (int e = 0; e < 8; ++e) acc += f[e] * qj[e];
  }
  part[t] = acc;
  __syncthreads();
  if (t < 8) {
    float sum = 0.f;
    #pragma unroll
    for (int g = 0; g < 8; ++g) sum += part[g * 8 + t];
    qsv[bi * 8 + t] += sum;
  }
}

// Fo[b,j,e] = sum_{i,s} Fso[b,i,j,s,e] * qs[b,i,s]; accumulate into qov.
__global__ void __launch_bounds__(64)
reduce_Fo(const float* __restrict__ Fso, const float* __restrict__ qs,
          float* __restrict__ qov)
{
  const int bj = blockIdx.x;          // b*NE + j
  const int b  = bj / NE;
  const int j  = bj - b * NE;
  const int t  = threadIdx.x;
  const int e  = t & 7;
  const int ig = t >> 3;
  __shared__ float part[64];
  float acc = 0.f;
  for (int i = ig; i < NE; i += 8) {
    const float* f  = Fso + ((size_t)(b * NE + i) * NE + j) * 64 + e;
    const float* qi = qs + (b * NE + i) * 8;
    #pragma unroll
    for (int s = 0; s < 8; ++s) acc += f[s * 8] * qi[s];
  }
  part[t] = acc;
  __syncthreads();
  if (t < 8) {
    float sum = 0.f;
    #pragma unroll
    for (int g = 0; g < 8; ++g) sum += part[g * 8 + t];
    qov[bj * 8 + t] += sum;
  }
}

// ---------------------------------------------------------------------------
extern "C" void kernel_launch(void* const* d_in, const int* in_sizes, int n_in,
                              void* d_out, int out_size, void* d_ws, size_t ws_size,
                              hipStream_t stream) {
  (void)in_sizes; (void)n_in; (void)out_size; (void)ws_size;
  const float* sub   = (const float*)d_in[0];
  const float* obj   = (const float*)d_in[1];
  const float* rel   = (const float*)d_in[2];
  const float* subsc = (const float*)d_in[3];
  const float* objsc = (const float*)d_in[4];
  const float* relsc = (const float*)d_in[5];
  const float* W1 = (const float*)d_in[6];  const float* b1 = (const float*)d_in[7];
  const float* W2 = (const float*)d_in[8];  const float* b2 = (const float*)d_in[9];
  const float* W3 = (const float*)d_in[10]; const float* b3 = (const float*)d_in[11];
  const float* We = (const float*)d_in[12]; const float* be = (const float*)d_in[13];
  const int*   ent = (const int*)d_in[14];

  char* w = (char*)d_ws;
  size_t off = 0;
  auto alloc = [&](size_t bytes) -> void* {
    void* p = w + off;
    off = (off + bytes + 255) & ~(size_t)255;
    return p;
  };
  const size_t PACKW = (size_t)(768 / 32) * 768 * 16;   // u32 count
  uint32_t* W1p = (uint32_t*)alloc(PACKW * 4);
  uint32_t* W2p = (uint32_t*)alloc(PACKW * 4);
  uint32_t* W3p = (uint32_t*)alloc(PACKW * 4);
  uint32_t* Wep = (uint32_t*)alloc(PACKW * 4);
  float*    L1  = (float*)   alloc((size_t)M1 * R * 4);
  float*    L2  = (float*)   alloc((size_t)M1 * R * 4);
  uint16_t* T   = (uint16_t*)alloc((size_t)M2 * R * 2);
  float*    Jm  = (float*)   alloc((size_t)M2 * R * 4);
  float*    qsb = (float*)   alloc((size_t)M1 * RE * 4);
  float*    qob = (float*)   alloc((size_t)M1 * RE * 4);
  float*    qrb = (float*)   alloc((size_t)M2 * RR * 4);
  float*    Fso = (float*)   alloc((size_t)M2 * 64 * 4);

  float* qsv = (float*)d_out;           // (BS,NE,RE)
  float* qov = qsv + M1 * RE;           // (BS,NE,RE)
  float* qrv = qov + M1 * RE;           // (BS,NE,NE,RR)

  // Pack weights into WMMA B-fragment layout (lives in L2: ~4.7 MB total).
  {
    int blocks = (int)((PACKW + 255) / 256);
    pack_w<<<blocks, 256, 0, stream>>>(W1, W1p);
    pack_w<<<blocks, 256, 0, stream>>>(W2, W2p);
    pack_w<<<blocks, 256, 0, stream>>>(W3, W3p);
    pack_w<<<blocks, 256, 0, stream>>>(We, Wep);
  }
  // L1 = sub@W1+b1, L2 = obj@W2+b2 (f32 out)
  gemm_wmma<0, true><<<dim3(M1 / 64, R / 64), 128, 0, stream>>>(
      sub, W1p, b1, L1, nullptr, nullptr, nullptr, nullptr, D, R);
  gemm_wmma<0, true><<<dim3(M1 / 64, R / 64), 128, 0, stream>>>(
      obj, W2p, b2, L2, nullptr, nullptr, nullptr, nullptr, D, R);
  // T = (rel@W3+b3) * l1 * l2 -> bf16 (l3 never materialized)
  gemm_wmma<1, true><<<dim3(M2 / 64, R / 64), 128, 0, stream>>>(
      rel, W3p, b3, nullptr, T, L1, L2, nullptr, D, R);
  // J = (T@We+be) * mask2d -> f32
  gemm_wmma<2, false><<<dim3(M2 / 64, R / 64), 128, 0, stream>>>(
      T, Wep, be, Jm, nullptr, nullptr, nullptr, ent, R, R);

  hipMemcpyAsync(qsv, subsc, (size_t)M1 * RE * 4, hipMemcpyDeviceToDevice, stream);
  hipMemcpyAsync(qov, objsc, (size_t)M1 * RE * 4, hipMemcpyDeviceToDevice, stream);
  hipMemcpyAsync(qrv, relsc, (size_t)M2 * RR * 4, hipMemcpyDeviceToDevice, stream);

  for (int it = 0; it < ITERS; ++it) {
    mask1d_k<<<(M1 * RE + 255) / 256, 256, 0, stream>>>(qsv, ent);
    mask1d_k<<<(M1 * RE + 255) / 256, 256, 0, stream>>>(qov, ent);
    mask2d_k<<<(M2 * RR + 255) / 256, 256, 0, stream>>>(qrv, ent);
    softmax_rows<RE><<<(M1 + 63) / 64, 64, 0, stream>>>(qsv, qsb, M1);
    softmax_rows<RE><<<(M1 + 63) / 64, 64, 0, stream>>>(qov, qob, M1);
    softmax_rows<RR><<<(M2 + 63) / 64, 64, 0, stream>>>(qrv, qrb, M2);
    pair_kernel<<<M2, 64, 0, stream>>>(Jm, qsb, qob, qrb, Fso, qrv);
    reduce_Fs<<<M1, 64, 0, stream>>>(Fso, qob, qsv);
    reduce_Fo<<<M1, 64, 0, stream>>>(Fso, qsb, qov);
  }
}
